// LSTM_56281251446920
// MI455X (gfx1250) — matrix operand order repaired and checked by reading, not compiled, not run on hardware
//
#include <hip/hip_runtime.h>
#include <hip/hip_bf16.h>

// ---------------------------------------------------------------------------
// Persistent LSTM kernel for MI455X (gfx1250, wave32, WMMA).
//   gates = x_t @ Wx + h_{t-1} @ Wh + b       (bf16 WMMA, f32 accum)
//   c = tanh(g)*sig(i) + c*sig(f);  h = tanh(c)*sig(o)
// 32 workgroups, each owning a 16-column slice of H.
//   - Weight slices converted once to bf16, resident in LDS (padded rows).
//   - Per step: x_t (f32->bf16) and h (bf16) staged cooperatively into LDS,
//     then the WMMA loops are pure ds_load_b128 + v_wmma.
//   - h exchanged between WGs in bf16 via a double-buffered global buffer.
//   - Device-wide step barrier: release/acquire atomic counter.
// ---------------------------------------------------------------------------

typedef __attribute__((ext_vector_type(16))) __bf16 v16bf;
typedef __attribute__((ext_vector_type(8)))  float  v8f;

namespace {
constexpr int kT       = 2048;
constexpr int kB       = 32;
constexpr int kE       = 512;
constexpr int kH       = 512;
constexpr int kNWG     = 32;    // kH / 16 column slices
constexpr int kThreads = 256;   // 8 wave32
constexpr int kLdsPitch = 520;  // 512 + 8 bf16 pad -> +4 bank rotation per row
}

union Frag {
  v16bf  v;
  uint4  u[2];
  __bf16 e[16];
};
union Pack8 {
  __bf16 e[8];
  uint4  u;
};

// A fragment (16x32 bf16) from LDS, rows padded to kLdsPitch.
// CDNA5 layout: lanes 0-15 -> rows m0+lane, K = k0+{0..7} then k0+{16..23};
//               lanes 16-31 -> rows m0+lane-16, K = k0+{8..15} then k0+{24..31}.
__device__ __forceinline__ v16bf a_frag_lds(const __bf16* m, int m0, int k0, int lane) {
  const int lr   = lane & 15;
  const int koff = (lane >> 4) << 3;  // 0 or 8
  const __bf16* p = m + (size_t)(m0 + lr) * kLdsPitch + k0 + koff;
  Frag f;
  f.u[0] = *(const uint4*)(p);
  f.u[1] = *(const uint4*)(p + 16);
  return f.v;
}

// B fragment (32x16 bf16) from LDS weight slice [64 cols][kLdsPitch k].
// CDNA5 layout: N = lane%16, K = k0 + (lane/16)*8 + {0..7} then k0 + 16 + (lane/16)*8 + {0..7}.
__device__ __forceinline__ v16bf b_frag_lds(const __bf16* w, int gate, int k0, int lane) {
  const __bf16* p = w + (size_t)(gate * 16 + (lane & 15)) * kLdsPitch + k0 + ((lane >> 4) << 3);
  Frag f;
  f.u[0] = *(const uint4*)(p);
  f.u[1] = *(const uint4*)(p + 16);
  return f.v;
}

__device__ __forceinline__ float sigm_fast(float x) {
  return 1.0f / (1.0f + __expf(-x));
}
__device__ __forceinline__ float tanh_fast(float x) {
  return 2.0f * sigm_fast(2.0f * x) - 1.0f;
}

__global__ __launch_bounds__(kThreads)
void lstm_persistent(const float* __restrict__ embeds,
                     const float* __restrict__ h0,
                     const float* __restrict__ c0,
                     const float* __restrict__ Wgx, const float* __restrict__ Wgh, const float* __restrict__ bg,
                     const float* __restrict__ Wix, const float* __restrict__ Wih, const float* __restrict__ bi,
                     const float* __restrict__ Wfx, const float* __restrict__ Wfh, const float* __restrict__ bf,
                     const float* __restrict__ Wox, const float* __restrict__ Woh, const float* __restrict__ bo,
                     float* __restrict__ out_seq,   // [T,B,H]
                     float* __restrict__ out_hT,    // [B,H]
                     float* __restrict__ out_cT,    // [B,H]
                     unsigned* __restrict__ sync_counter,
                     __bf16* __restrict__ hbuf)     // [2][B][H] bf16
{
  __shared__ __bf16 sWx[4 * 16 * kLdsPitch];   // 65 KB  x-side weights [gate*16+n][k]
  __shared__ __bf16 sWh[4 * 16 * kLdsPitch];   // 65 KB  h-side weights
  __shared__ __bf16 sX[kB * kLdsPitch];        // 32.5 KB staged x_t (bf16)
  __shared__ __bf16 sH[kB * kLdsPitch];        // 32.5 KB staged h_{t-1} (bf16)
  __shared__ float  sGates[2][4][16][17];      // (mt, gate, row, col) +pad
  __shared__ float  sC[32][17];                // cell state slice (+pad)

  const int tid  = threadIdx.x;
  const int lane = tid & 31;
  const int wave = tid >> 5;
  const int mt   = wave & 1;     // batch-row tile: rows [mt*16, mt*16+16)
  const int gate = wave >> 1;    // 0=g 1=i 2=f 3=o
  const int wg   = blockIdx.x;   // H column slice
  const int col0 = wg * 16;

  // staging coords: 8 threads per batch row, 64 elements each
  const int sm  = tid >> 3;        // 0..31 (batch row)
  const int sk0 = (tid & 7) * 64;  // 0..448

  // ---- One-time: stage f32 weight slices into LDS as bf16 -----------------
  {
    const float* Wxp[4] = {Wgx, Wix, Wfx, Wox};
    const float* Whp[4] = {Wgh, Wih, Wfh, Woh};
    const int kk = tid >> 4;     // 0..15
    const int nn = tid & 15;     // 0..15
    for (int g = 0; g < 4; ++g) {
      const float* wx = Wxp[g];
      const float* wh = Whp[g];
      for (int p = 0; p < 512 / 16; ++p) {
        const int k = p * 16 + kk;
        sWx[(size_t)(g * 16 + nn) * kLdsPitch + k] = (__bf16)wx[(size_t)k * kH + col0 + nn];
        sWh[(size_t)(g * 16 + nn) * kLdsPitch + k] = (__bf16)wh[(size_t)k * kH + col0 + nn];
      }
    }
    // init cell state slice from c0
    for (int e = tid; e < kB * 16; e += kThreads) {
      const int m = e >> 4, cc = e & 15;
      sC[m][cc] = c0[(size_t)m * kH + col0 + cc];
    }
  }

  // per-lane gate bias for this wave's output column
  const float* bp[4] = {bg, bi, bf, bo};
  const float bias = bp[gate][col0 + (lane & 15)];

  const int m0 = mt * 16;

  for (int t = 0; t < kT; ++t) {
    const float* xbase = embeds + (size_t)t * kB * kE;
    __bf16* hwrite = hbuf + (size_t)((t + 1) & 1) * kB * kH;

    // prefetch next timestep's x block (64KB, L2-resident thereafter)
    if (t + 1 < kT)
      __builtin_prefetch(embeds + (size_t)(t + 1) * kB * kE + (size_t)tid * 256, 0, 1);

    // ---- stage x_t (f32 -> bf16) and h_{t-1} into LDS ---------------------
    {
      const float* xs = xbase + (size_t)sm * kE + sk0;
      __bf16* xd = &sX[(size_t)sm * kLdsPitch + sk0];
#pragma unroll
      for (int j = 0; j < 64; j += 8) {
        float4 f0 = *(const float4*)(xs + j);
        float4 f1 = *(const float4*)(xs + j + 4);
        Pack8 pk;
        pk.e[0] = (__bf16)f0.x; pk.e[1] = (__bf16)f0.y;
        pk.e[2] = (__bf16)f0.z; pk.e[3] = (__bf16)f0.w;
        pk.e[4] = (__bf16)f1.x; pk.e[5] = (__bf16)f1.y;
        pk.e[6] = (__bf16)f1.z; pk.e[7] = (__bf16)f1.w;
        *(uint4*)(xd + j) = pk.u;
      }
      __bf16* hd = &sH[(size_t)sm * kLdsPitch + sk0];
      if (t == 0) {
        const float* hs = h0 + (size_t)sm * kH + sk0;
#pragma unroll
        for (int j = 0; j < 64; j += 8) {
          float4 f0 = *(const float4*)(hs + j);
          float4 f1 = *(const float4*)(hs + j + 4);
          Pack8 pk;
          pk.e[0] = (__bf16)f0.x; pk.e[1] = (__bf16)f0.y;
          pk.e[2] = (__bf16)f0.z; pk.e[3] = (__bf16)f0.w;
          pk.e[4] = (__bf16)f1.x; pk.e[5] = (__bf16)f1.y;
          pk.e[6] = (__bf16)f1.z; pk.e[7] = (__bf16)f1.w;
          *(uint4*)(hd + j) = pk.u;
        }
      } else {
        const __bf16* hs = hbuf + (size_t)(t & 1) * kB * kH + (size_t)sm * kH + sk0;
#pragma unroll
        for (int j = 0; j < 64; j += 8)
          *(uint4*)(hd + j) = *(const uint4*)(hs + j);
      }
    }
    __syncthreads();

    // ---- GEMM: all-LDS WMMA loops ----------------------------------------
    v8f acc = {};
#pragma unroll 4
    for (int ks = 0; ks < kE / 32; ++ks) {    // x_t @ Wx
      v16bf a = a_frag_lds(sX, m0, ks * 32, lane);
      v16bf b = b_frag_lds(sWx, gate, ks * 32, lane);
      acc = __builtin_amdgcn_wmma_f32_16x16x32_bf16(false, a, false, b,
                                                    (short)0, acc, false, false);
    }
#pragma unroll 4
    for (int ks = 0; ks < kH / 32; ++ks) {    // h_{t-1} @ Wh
      v16bf a = a_frag_lds(sH, m0, ks * 32, lane);
      v16bf b = b_frag_lds(sWh, gate, ks * 32, lane);
      acc = __builtin_amdgcn_wmma_f32_16x16x32_bf16(false, a, false, b,
                                                    (short)0, acc, false, false);
    }

    // C/D layout: VGPR r -> row r + 8*(lane/16), col lane%16
#pragma unroll
    for (int r = 0; r < 8; ++r) {
      const int row = r + ((lane >> 4) << 3);
      sGates[mt][gate][row][lane & 15] = acc[r] + bias;
    }
    __syncthreads();

    // ---- Elementwise LSTM cell update for this WG's 32x16 slice ----------
    for (int e = tid; e < kB * 16; e += kThreads) {
      const int m  = e >> 4;
      const int cc = e & 15;
      const int rt = m >> 4, rr = m & 15;
      const float gv = tanh_fast(sGates[rt][0][rr][cc]);
      const float iv = sigm_fast(sGates[rt][1][rr][cc]);
      const float fv = sigm_fast(sGates[rt][2][rr][cc]);
      const float ov = sigm_fast(sGates[rt][3][rr][cc]);
      const float cprev = sC[m][cc];
      const float cnew  = gv * iv + cprev * fv;
      const float hv    = tanh_fast(cnew) * ov;
      sC[m][cc] = cnew;
      const int col = col0 + cc;
      hwrite[(size_t)m * kH + col] = (__bf16)hv;
      out_seq[((size_t)t * kB + m) * kH + col] = hv;
      if (t == kT - 1) {
        out_hT[(size_t)m * kH + col] = hv;
        out_cT[(size_t)m * kH + col] = cnew;
      }
    }

    // ---- device-wide step barrier (double-buffered h: 1 sync per step) ----
    __threadfence();
    __syncthreads();
    if (tid == 0) {
      __hip_atomic_fetch_add(sync_counter, 1u, __ATOMIC_RELEASE,
                             __HIP_MEMORY_SCOPE_AGENT);
      const unsigned target = (unsigned)(t + 1) * kNWG;
      while (__hip_atomic_load(sync_counter, __ATOMIC_ACQUIRE,
                               __HIP_MEMORY_SCOPE_AGENT) < target) {
        __builtin_amdgcn_s_sleep(2);
      }
    }
    __syncthreads();
  }
}

extern "C" void kernel_launch(void* const* d_in, const int* in_sizes, int n_in,
                              void* d_out, int out_size, void* d_ws, size_t ws_size,
                              hipStream_t stream) {
  (void)in_sizes; (void)n_in; (void)out_size; (void)ws_size;

  const float* embeds = (const float*)d_in[0];
  const float* h0     = (const float*)d_in[1];
  const float* c0     = (const float*)d_in[2];
  const float* Wgx = (const float*)d_in[3];
  const float* Wgh = (const float*)d_in[4];
  const float* bg  = (const float*)d_in[5];
  const float* Wix = (const float*)d_in[6];
  const float* Wih = (const float*)d_in[7];
  const float* bi  = (const float*)d_in[8];
  const float* Wfx = (const float*)d_in[9];
  const float* Wfh = (const float*)d_in[10];
  const float* bf  = (const float*)d_in[11];
  const float* Wox = (const float*)d_in[12];
  const float* Woh = (const float*)d_in[13];
  const float* bo  = (const float*)d_in[14];

  float* out_seq = (float*)d_out;                               // [T,B,H]
  float* out_hT  = out_seq + (size_t)kT * kB * kH;              // [B,H]
  float* out_cT  = out_hT + (size_t)kB * kH;                    // [B,H]

  unsigned* sync_counter = (unsigned*)d_ws;                     // reset below
  __bf16*   hbuf         = (__bf16*)((char*)d_ws + 256);        // 2*[B,H] bf16 = 64 KB

  // reset the step-barrier counter (graph-capture safe)
  hipMemsetAsync(d_ws, 0, 256, stream);

  lstm_persistent<<<dim3(kNWG), dim3(kThreads), 0, stream>>>(
      embeds, h0, c0,
      Wgx, Wgh, bg, Wix, Wih, bi, Wfx, Wfh, bf, Wox, Woh, bo,
      out_seq, out_hT, out_cT, sync_counter, hbuf);
}
